// Joint_72146860638297
// MI455X (gfx1250) — matrix operand order repaired and checked by reading, not compiled, not run on hardware
//
#include <hip/hip_runtime.h>
#include <hip/hip_bf16.h>
#include <math.h>

typedef _Float16 f16;
typedef __attribute__((ext_vector_type(8)))  _Float16 v8h;
typedef __attribute__((ext_vector_type(16))) _Float16 v16h;
typedef __attribute__((ext_vector_type(8)))  float    v8f;

#define IN_C  768
#define HID   568
#define HIDP  640      // HID padded to a multiple of 128 (tile N) / 32 (WMMA K)
#define OUT_C 1024
#define SEQ   2048
#define NB    4
#define NTOK  (NB * SEQ)   // 8192

// ---------------------------------------------------------------------------
// f32 -> f16 cast (grid-stride)
// ---------------------------------------------------------------------------
__global__ __launch_bounds__(256) void cast_f16_kernel(
    const float* __restrict__ in, f16* __restrict__ out, long n) {
  long i = (long)blockIdx.x * 256 + threadIdx.x;
  const long stride = (long)gridDim.x * 256;
  for (; i < n; i += stride) out[i] = (f16)in[i];
}

// ---------------------------------------------------------------------------
// Tiled WMMA GEMM:  C = act(alpha * A @ B(+T) + bias)
//   A: M x K (f16, row-major, lda)
//   B: NN -> K x N row-major (ldb; cols >= nvalid read as 0)
//      NT -> N x K row-major (ldb)        [Q @ K^T]
//   C: M x N as f32 (Cf) and/or f16 (Ch), ldc
// 128 threads = 4 waves (2x2); block tile 128x128; wave tile 64x64.
// Per k-step: 16 ds_load_b128 : 16 v_wmma (1:1). Double-buffered LDS,
// one barrier per k-step. Staging kept in v8h registers (no union ->
// no scratch); A-fragment streamed per mi to stay under 256 VGPRs.
// ---------------------------------------------------------------------------
template <bool TRANS_B>
__global__ __launch_bounds__(128, 1) void gemm_wmma(
    const f16* __restrict__ A, int lda, long strideA,
    const f16* __restrict__ B, int ldb, long strideB,
    const float* __restrict__ bias,
    float* __restrict__ Cf, f16* __restrict__ Ch, int ldc, long strideC,
    int M, int K, int nvalid, int do_relu, float alpha) {
  constexpr int BM = 128, BN = 128, BK = 32, BKP = BK + 8;
  __shared__ f16 sA[2][BM][BKP];   // rows 16B-aligned (80B stride)
  __shared__ f16 sB[2][BN][BKP];   // [n][k]: fragment reads contiguous in k

  const int tid  = threadIdx.x;    // 0..127
  const int lane = tid & 31;       // wave32
  const int wave = tid >> 5;       // 0..3
  const long zb  = blockIdx.z;
  const f16* Ab = A + zb * strideA;
  const f16* Bb = B + zb * strideB;

  const int bm = blockIdx.y * BM;
  const int bn = blockIdx.x * BN;
  const int m0 = (wave >> 1) * 64;
  const int n0 = (wave & 1) * 64;

  v8f acc[4][4] = {};

  // B (NN) staging assignment: thread -> (k-row, 32-wide n segment)
  const int bk  = tid >> 2;          // 0..31
  const int bns = (tid & 3) * 32;    // 0,32,64,96
  // Uniform edge test: only the last block-column of the h-GEMM is partial.
  const bool tile_full = TRANS_B || (bn + BN <= nvalid);

  // ISA fragment K-bases (16-bit A 16x32 / B 32x16 layouts)
  const int kbA = (lane >= 16) ? 8 : 0;
  const int kbB = (lane >= 16) ? 8 : 0;

  // Register staging (plain ext-vectors: SROA-friendly, no scratch)
  v8h a0, a1, a2, a3, b0, b1, b2, b3;

  auto load_tile = [&](int k0) {
    const f16* ga = Ab + (long)(bm + tid) * lda + k0;
    a0 = *(const v8h*)(ga);
    a1 = *(const v8h*)(ga + 8);
    a2 = *(const v8h*)(ga + 16);
    a3 = *(const v8h*)(ga + 24);
    if (TRANS_B) {
      const f16* gb = Bb + (long)(bn + tid) * ldb + k0;
      b0 = *(const v8h*)(gb);
      b1 = *(const v8h*)(gb + 8);
      b2 = *(const v8h*)(gb + 16);
      b3 = *(const v8h*)(gb + 24);
    } else {
      const f16* gb = Bb + (long)(k0 + bk) * ldb + (bn + bns);
      if (tile_full) {
        b0 = *(const v8h*)(gb);
        b1 = *(const v8h*)(gb + 8);
        b2 = *(const v8h*)(gb + 16);
        b3 = *(const v8h*)(gb + 24);
      } else {
#pragma unroll
        for (int i = 0; i < 8; i++) {
          b0[i] = (bn + bns + i      < nvalid) ? gb[i]      : (f16)0.0f;
          b1[i] = (bn + bns + i + 8  < nvalid) ? gb[i + 8]  : (f16)0.0f;
          b2[i] = (bn + bns + i + 16 < nvalid) ? gb[i + 16] : (f16)0.0f;
          b3[i] = (bn + bns + i + 24 < nvalid) ? gb[i + 24] : (f16)0.0f;
        }
      }
    }
  };

  auto store_tile = [&](int buf) {
    *(v8h*)(&sA[buf][tid][0])  = a0;
    *(v8h*)(&sA[buf][tid][8])  = a1;
    *(v8h*)(&sA[buf][tid][16]) = a2;
    *(v8h*)(&sA[buf][tid][24]) = a3;
    if (TRANS_B) {
      *(v8h*)(&sB[buf][tid][0])  = b0;
      *(v8h*)(&sB[buf][tid][8])  = b1;
      *(v8h*)(&sB[buf][tid][16]) = b2;
      *(v8h*)(&sB[buf][tid][24]) = b3;
    } else {
#pragma unroll
      for (int i = 0; i < 8; i++) {
        sB[buf][bns + i][bk]      = b0[i];
        sB[buf][bns + i + 8][bk]  = b1[i];
        sB[buf][bns + i + 16][bk] = b2[i];
        sB[buf][bns + i + 24][bk] = b3[i];
      }
    }
  };

  const int nt = K / BK;
  load_tile(0);

  int buf = 0;
  for (int t = 0; t < nt; ++t) {
    store_tile(buf);
    __syncthreads();

    // issue global loads for tile t+1 (overlaps WMMA stream below)
    if (t + 1 < nt) load_tile((t + 1) * BK);

    // B fragments resident (4 x 8 VGPRs); A fragment streamed per mi
    v16h bfr[4];
#pragma unroll
    for (int ni = 0; ni < 4; ni++) {
      const f16* pb = &sB[buf][n0 + ni * 16 + (lane & 15)][kbB];
      v8h lo = *(const v8h*)(pb);        // K = kbB .. kbB+7
      v8h hi = *(const v8h*)(pb + 16);   // K = kbB+16 .. kbB+23
      bfr[ni] = __builtin_shufflevector(lo, hi, 0, 1, 2, 3, 4, 5, 6, 7, 8, 9,
                                        10, 11, 12, 13, 14, 15);
    }
#pragma unroll
    for (int mi = 0; mi < 4; mi++) {
      const f16* pa = &sA[buf][m0 + mi * 16 + (lane & 15)][kbA];
      v8h lo = *(const v8h*)(pa);        // K = kbA .. kbA+7
      v8h hi = *(const v8h*)(pa + 16);   // K = kbA+16 .. kbA+23
      v16h afr = __builtin_shufflevector(lo, hi, 0, 1, 2, 3, 4, 5, 6, 7, 8, 9,
                                         10, 11, 12, 13, 14, 15);
#pragma unroll
      for (int ni = 0; ni < 4; ni++)
        acc[mi][ni] = __builtin_amdgcn_wmma_f32_16x16x32_f16(
            false, afr, false, bfr[ni], (short)0, acc[mi][ni], false, false);
    }

    buf ^= 1;
  }

  // ---- epilogue: scale, bias, relu, store (16x16 f32 C/D layout) ----
  const int mhi  = (lane >= 16) ? 8 : 0;
  const int ncol = lane & 15;
#pragma unroll
  for (int ni = 0; ni < 4; ni++) {
    const int gn = bn + n0 + ni * 16 + ncol;
    const float bv = (bias && gn < nvalid) ? bias[gn] : 0.0f;
#pragma unroll
    for (int mi = 0; mi < 4; mi++) {
#pragma unroll
      for (int r = 0; r < 8; r++) {
        const long gm = bm + m0 + mi * 16 + r + mhi;
        float v = acc[mi][ni][r] * alpha + bv;
        if (do_relu) v = fmaxf(v, 0.0f);
        const long idx = zb * strideC + gm * (long)ldc + gn;
        if (Cf) Cf[idx] = v;
        if (Ch) Ch[idx] = (f16)v;
      }
    }
  }
}

// ---------------------------------------------------------------------------
// Row softmax: f32 scores row (2048) -> f16 probabilities. One block per row.
// ---------------------------------------------------------------------------
__global__ __launch_bounds__(256) void softmax_rows(
    const float* __restrict__ S, f16* __restrict__ P) {
  const int N = SEQ;  // 2048 = 8 * 256
  const long row = blockIdx.x;
  const float* s = S + row * (long)N;
  f16* p = P + row * (long)N;
  const int tid = threadIdx.x;
  __shared__ float red[8];

  float loc[8];
  float vmax = -3.0e38f;
#pragma unroll
  for (int j = 0; j < 8; j++) {
    loc[j] = s[tid + j * 256];
    vmax = fmaxf(vmax, loc[j]);
  }
  for (int o = 16; o > 0; o >>= 1) vmax = fmaxf(vmax, __shfl_xor(vmax, o, 32));
  if ((tid & 31) == 0) red[tid >> 5] = vmax;
  __syncthreads();
  vmax = red[0];
#pragma unroll
  for (int i = 1; i < 8; i++) vmax = fmaxf(vmax, red[i]);
  __syncthreads();

  float sum = 0.0f;
#pragma unroll
  for (int j = 0; j < 8; j++) {
    loc[j] = __expf(loc[j] - vmax);
    sum += loc[j];
  }
  for (int o = 16; o > 0; o >>= 1) sum += __shfl_xor(sum, o, 32);
  if ((tid & 31) == 0) red[tid >> 5] = sum;
  __syncthreads();
  sum = 0.0f;
#pragma unroll
  for (int i = 0; i < 8; i++) sum += red[i];
  const float rinv = 1.0f / sum;
#pragma unroll
  for (int j = 0; j < 8; j++) p[tid + j * 256] = (f16)(loc[j] * rinv);
}

// ---------------------------------------------------------------------------
// y = LN(a + b) * g + be  [optional relu]; writes f32 and/or f16.
// One block per 1024-wide row. Matches jnp.var (biased, /N), eps=1e-5.
// ---------------------------------------------------------------------------
__global__ __launch_bounds__(256) void add_ln_kernel(
    const float* __restrict__ A, const float* __restrict__ B,
    const float* __restrict__ g, const float* __restrict__ be,
    float* __restrict__ outf, f16* __restrict__ outh, int do_relu) {
  const int C = OUT_C;  // 1024 = 4 * 256
  const long row = blockIdx.x;
  const int tid = threadIdx.x;
  const float* a = A + row * (long)C;
  const float* b = B + row * (long)C;
  __shared__ float r1[8], r2[8];

  float u[4];
  float s = 0.0f, sq = 0.0f;
#pragma unroll
  for (int j = 0; j < 4; j++) {
    const float v = a[tid + j * 256] + b[tid + j * 256];
    u[j] = v;
    s += v;
    sq += v * v;
  }
  for (int o = 16; o > 0; o >>= 1) {
    s += __shfl_xor(s, o, 32);
    sq += __shfl_xor(sq, o, 32);
  }
  if ((tid & 31) == 0) { r1[tid >> 5] = s; r2[tid >> 5] = sq; }
  __syncthreads();
  s = 0.0f; sq = 0.0f;
#pragma unroll
  for (int i = 0; i < 8; i++) { s += r1[i]; sq += r2[i]; }
  const float mean = s * (1.0f / C);
  const float var  = sq * (1.0f / C) - mean * mean;
  const float rstd = rsqrtf(var + 1e-5f);
#pragma unroll
  for (int j = 0; j < 4; j++) {
    const int i = tid + j * 256;
    float y = (u[j] - mean) * rstd * g[i] + be[i];
    if (do_relu) y = fmaxf(y, 0.0f);
    if (outf) outf[row * (long)C + i] = y;
    if (outh) outh[row * (long)C + i] = (f16)y;
  }
}

// ---------------------------------------------------------------------------
// Host-side orchestration
// ---------------------------------------------------------------------------
extern "C" void kernel_launch(void* const* d_in, const int* in_sizes, int n_in,
                              void* d_out, int out_size, void* d_ws,
                              size_t ws_size, hipStream_t stream) {
  (void)in_sizes; (void)n_in; (void)out_size; (void)ws_size;

  const float* x     = (const float*)d_in[0];
  const float* w_mlp = (const float*)d_in[1];
  const float* b_mlp = (const float*)d_in[2];
  const float* wq    = (const float*)d_in[3];
  const float* bq    = (const float*)d_in[4];
  const float* wk    = (const float*)d_in[5];
  const float* bk    = (const float*)d_in[6];
  const float* wv    = (const float*)d_in[7];
  const float* bv    = (const float*)d_in[8];
  const float* wm    = (const float*)d_in[9];
  const float* bm    = (const float*)d_in[10];
  const float* g1    = (const float*)d_in[11];
  const float* be1   = (const float*)d_in[12];
  const float* wf1   = (const float*)d_in[13];
  const float* bf1   = (const float*)d_in[14];
  const float* wf2   = (const float*)d_in[15];
  const float* bf2   = (const float*)d_in[16];
  const float* g2    = (const float*)d_in[17];
  const float* be2   = (const float*)d_in[18];

  char* ws = (char*)d_ws;
  size_t off = 0;
  auto alloc = [&](size_t bytes) {
    size_t o = off;
    off += (bytes + 255) & ~(size_t)255;
    return o;
  };

  const size_t oXH   = alloc((size_t)NTOK * IN_C * 2);
  const size_t oWMLP = alloc((size_t)IN_C * HID * 2);
  const size_t oWQ   = alloc((size_t)HID * OUT_C * 2);
  const size_t oWK   = alloc((size_t)HID * OUT_C * 2);
  const size_t oWV   = alloc((size_t)HID * OUT_C * 2);
  const size_t oWM   = alloc((size_t)HID * OUT_C * 2);
  const size_t oWF1  = alloc((size_t)OUT_C * OUT_C * 2);
  const size_t oWF2  = alloc((size_t)OUT_C * OUT_C * 2);
  const size_t oH    = alloc((size_t)NTOK * HIDP * 2);
  const size_t oQ    = alloc((size_t)NTOK * OUT_C * 2);
  const size_t oK    = alloc((size_t)NTOK * OUT_C * 2);  (void)oK;
  const size_t oV    = alloc((size_t)NTOK * OUT_C * 2);
  const size_t oXMOD = alloc((size_t)NTOK * OUT_C * 4);
  const size_t oS    = alloc((size_t)NB * SEQ * SEQ * 4);
  const size_t oP    = alloc((size_t)NB * SEQ * SEQ * 2);
  // Lifetime-based aliases (written strictly after their donors are dead):
  const size_t oAO   = oS;                                 // attn_out (f32)
  const size_t oFFN2 = oS + (size_t)NTOK * OUT_C * 4;      // ffn2 (f32)
  const size_t oX1H  = oP;                                 // x1 (f16)
  const size_t oFF1H = oP + (size_t)NTOK * OUT_C * 2;      // relu(ffn1) (f16)
  const size_t oX1F  = oQ;                                 // x1 (f32) over Q+K

  f16*   XH   = (f16*)(ws + oXH);
  f16*   WMLP = (f16*)(ws + oWMLP);
  f16*   WQ   = (f16*)(ws + oWQ);
  f16*   WK   = (f16*)(ws + oWK);
  f16*   WV   = (f16*)(ws + oWV);
  f16*   WM   = (f16*)(ws + oWM);
  f16*   WF1  = (f16*)(ws + oWF1);
  f16*   WF2  = (f16*)(ws + oWF2);
  f16*   H    = (f16*)(ws + oH);
  f16*   Q    = (f16*)(ws + oQ);
  f16*   Kk   = (f16*)(ws + oK);
  f16*   V    = (f16*)(ws + oV);
  float* XMOD = (float*)(ws + oXMOD);
  float* S    = (float*)(ws + oS);
  f16*   P    = (f16*)(ws + oP);
  float* AO   = (float*)(ws + oAO);
  float* FFN2 = (float*)(ws + oFFN2);
  f16*   X1H  = (f16*)(ws + oX1H);
  f16*   FF1H = (f16*)(ws + oFF1H);
  float* X1F  = (float*)(ws + oX1F);

  const dim3 blk256(256);
  const dim3 blkG(128);   // GEMM blocks: 4 waves

  // ---- 0. cast activations + weights to f16 ----
  cast_f16_kernel<<<1024, blk256, 0, stream>>>(x, XH, (long)NTOK * IN_C);
  cast_f16_kernel<<<256, blk256, 0, stream>>>(w_mlp, WMLP, (long)IN_C * HID);
  cast_f16_kernel<<<256, blk256, 0, stream>>>(wq, WQ, (long)HID * OUT_C);
  cast_f16_kernel<<<256, blk256, 0, stream>>>(wk, WK, (long)HID * OUT_C);
  cast_f16_kernel<<<256, blk256, 0, stream>>>(wv, WV, (long)HID * OUT_C);
  cast_f16_kernel<<<256, blk256, 0, stream>>>(wm, WM, (long)HID * OUT_C);
  cast_f16_kernel<<<512, blk256, 0, stream>>>(wf1, WF1, (long)OUT_C * OUT_C);
  cast_f16_kernel<<<512, blk256, 0, stream>>>(wf2, WF2, (long)OUT_C * OUT_C);

  // ---- 1. h = relu(x @ w_mlp + b_mlp), stored padded to HIDP (zeros) ----
  gemm_wmma<false><<<dim3(HIDP / 128, NTOK / 128, 1), blkG, 0, stream>>>(
      XH, IN_C, 0, WMLP, HID, 0, b_mlp, nullptr, H, HIDP, 0,
      NTOK, IN_C, HID, 1, 1.0f);

  // ---- 2. q/k/v (f16) and x_mod (f32) projections, K = HIDP ----
  gemm_wmma<false><<<dim3(OUT_C / 128, NTOK / 128, 1), blkG, 0, stream>>>(
      H, HIDP, 0, WQ, OUT_C, 0, bq, nullptr, Q, OUT_C, 0,
      NTOK, HIDP, OUT_C, 0, 1.0f);
  gemm_wmma<false><<<dim3(OUT_C / 128, NTOK / 128, 1), blkG, 0, stream>>>(
      H, HIDP, 0, WK, OUT_C, 0, bk, nullptr, Kk, OUT_C, 0,
      NTOK, HIDP, OUT_C, 0, 1.0f);
  gemm_wmma<false><<<dim3(OUT_C / 128, NTOK / 128, 1), blkG, 0, stream>>>(
      H, HIDP, 0, WV, OUT_C, 0, bv, nullptr, V, OUT_C, 0,
      NTOK, HIDP, OUT_C, 0, 1.0f);
  gemm_wmma<false><<<dim3(OUT_C / 128, NTOK / 128, 1), blkG, 0, stream>>>(
      H, HIDP, 0, WM, OUT_C, 0, bm, XMOD, nullptr, OUT_C, 0,
      NTOK, HIDP, OUT_C, 0, 1.0f);

  // ---- 3. S = (q @ k^T) / 32, batched over NB ----
  gemm_wmma<true><<<dim3(SEQ / 128, SEQ / 128, NB), blkG, 0, stream>>>(
      Q, OUT_C, (long)SEQ * OUT_C, Kk, OUT_C, (long)SEQ * OUT_C, nullptr,
      S, nullptr, SEQ, (long)SEQ * SEQ, SEQ, OUT_C, SEQ, 0, 0.03125f);

  // ---- 4. P = softmax_rows(S) (f16) ----
  softmax_rows<<<NB * SEQ, blk256, 0, stream>>>(S, P);

  // ---- 5. attn_out = P @ v, batched ----
  gemm_wmma<false><<<dim3(OUT_C / 128, SEQ / 128, NB), blkG, 0, stream>>>(
      P, SEQ, (long)SEQ * SEQ, V, OUT_C, (long)SEQ * OUT_C, nullptr,
      AO, nullptr, OUT_C, (long)SEQ * OUT_C, SEQ, SEQ, OUT_C, 0, 1.0f);

  // ---- 6. x1 = LN(x_mod + attn_out) -> f32 + f16 ----
  add_ln_kernel<<<NTOK, blk256, 0, stream>>>(XMOD, AO, g1, be1, X1F, X1H, 0);

  // ---- 7. ffn = relu(x1 @ wf1 + bf1) @ wf2 + bf2 ----
  gemm_wmma<false><<<dim3(OUT_C / 128, NTOK / 128, 1), blkG, 0, stream>>>(
      X1H, OUT_C, 0, WF1, OUT_C, 0, bf1, nullptr, FF1H, OUT_C, 0,
      NTOK, OUT_C, OUT_C, 1, 1.0f);
  gemm_wmma<false><<<dim3(OUT_C / 128, NTOK / 128, 1), blkG, 0, stream>>>(
      FF1H, OUT_C, 0, WF2, OUT_C, 0, bf2, FFN2, nullptr, OUT_C, 0,
      NTOK, OUT_C, OUT_C, 0, 1.0f);

  // ---- 8. out = relu(LN(x1 + ffn)) ----
  add_ln_kernel<<<NTOK, blk256, 0, stream>>>(X1F, FFN2, g2, be2,
                                             (float*)d_out, nullptr, 1);
}
